// GCN_66374424592406
// MI455X (gfx1250) — compile-verified, matrix operand order checked by reading, not run on hardware
//
#include <hip/hip_runtime.h>
#include <hip/hip_bf16.h>

typedef __attribute__((ext_vector_type(16))) __bf16 v16bf;
typedef __attribute__((ext_vector_type(8)))  __bf16 v8bf;
typedef __attribute__((ext_vector_type(8)))  float  v8f;

#define BN_EPS 1e-5f

// ---------------------------------------------------------------- zero fill
__global__ void k_zero(float* __restrict__ p, long n) {
    long i = blockIdx.x * (long)blockDim.x + threadIdx.x;
    long stride = (long)gridDim.x * blockDim.x;
    for (; i < n; i += stride) p[i] = 0.0f;
}

// ------------------------------------------- f32 -> bf16 (optional row gather)
__global__ void k_cvt_rows(const float* __restrict__ src, const int* __restrict__ vert,
                           __bf16* __restrict__ dst, int nrows, int ncols) {
    long total = (long)nrows * ncols;
    long i = blockIdx.x * (long)blockDim.x + threadIdx.x;
    long stride = (long)gridDim.x * blockDim.x;
    for (; i < total; i += stride) {
        int r = (int)(i / ncols);
        int c = (int)(i - (long)r * ncols);
        int sr = vert ? vert[r] : r;
        dst[i] = (__bf16)src[(long)sr * ncols + c];
    }
}

// ------------------------------------------------- bf16 WMMA GEMM  C = A @ B
// A: [M,K] bf16 row-major, B: [K,N] bf16 row-major, C: [M,N] f32 row-major.
// N, K compile-time. One wave owns a full 16-row stripe (all N/16 tiles):
// A fragment loaded once per K-step, reused across all N-tiles. B is staged
// per-block into LDS in fragment-ready layout so each frag is one contiguous
// 32B ds read. v_wmma_f32_16x16x32_bf16, fully unrolled: (K/32)*(N/16) wmmas.
template <int N, int K>
__global__ void k_gemm_wmma(const __bf16* __restrict__ A, const __bf16* __restrict__ B,
                            float* __restrict__ C, int M) {
    constexpr int TN = N / 16;        // N-tiles per wave
    constexpr int KB = K / 32;        // K-steps
    __shared__ __bf16 sB[KB * TN * 32 * 16];   // fragment-ready B (K*N elems)

    const int lane = threadIdx.x & 31;
    const int l    = lane & 15;
    const int hi   = lane >> 4;       // 0 or 1

    // ---- cooperative fill: B[k][n] -> fragment layout [kb][tn][lane][v]
    for (int idx = threadIdx.x; idx < KB * TN * 32 * 16; idx += blockDim.x) {
        int v  = idx & 15;
        int ln = (idx >> 4) & 31;
        int t  = (idx >> 9) % TN;
        int kb = (idx >> 9) / TN;
        int i  = v >> 1, p = v & 1;
        int kk = kb * 32 + 2 * (i & 3) + ((i & 4) ? 16 : 0) + ((ln & 16) ? 8 : 0) + p;
        int cn = t * 16 + (ln & 15);
        sB[idx] = B[(size_t)kk * N + cn];
    }
    __syncthreads();

    const int tm = (int)((blockIdx.x * (long)blockDim.x + threadIdx.x) >> 5);
    const int tilesM = (M + 15) >> 4;
    if (tm < tilesM) {                // wave-uniform: EXEC all-ones inside
        int arow = tm * 16 + l;
        if (arow >= M) arow = M - 1;  // cold tail clamp
        const __bf16* ap = A + (size_t)arow * K + (hi ? 8 : 0);

        v8f acc[TN];
#pragma unroll
        for (int t = 0; t < TN; ++t) acc[t] = (v8f){};

#pragma unroll
        for (int kb = 0; kb < KB; ++kb) {
            v8bf alo = *(const v8bf*)(ap + kb * 32);
            v8bf ahi = *(const v8bf*)(ap + kb * 32 + 16);
            v16bf a = __builtin_shufflevector(alo, ahi, 0, 1, 2, 3, 4, 5, 6, 7,
                                              8, 9, 10, 11, 12, 13, 14, 15);
#pragma unroll
            for (int t = 0; t < TN; ++t) {
                v16bf b = *(const v16bf*)(&sB[((kb * TN + t) * 32 + lane) * 16]);
                acc[t] = __builtin_amdgcn_wmma_f32_16x16x32_bf16(
                             false, a, false, b, (short)0, acc[t], false, false);
            }
        }

        float* cp = C + (size_t)(tm * 16 + hi * 8) * N + l;
        if (tm * 16 + 16 <= M) {      // common path: immediate-offset stores
#pragma unroll
            for (int t = 0; t < TN; ++t)
#pragma unroll
                for (int r = 0; r < 8; ++r)
                    cp[(size_t)r * N + t * 16] = acc[t][r];
        } else {                      // cold tail
#pragma unroll
            for (int t = 0; t < TN; ++t)
#pragma unroll
                for (int r = 0; r < 8; ++r)
                    if (tm * 16 + hi * 8 + r < M) cp[(size_t)r * N + t * 16] = acc[t][r];
        }
    }
}

// --------------------------------------------- SpMM: Out[row] += val * S[col]
// One block per edge, one thread per feature. f32 global atomic add (support
// matrix is 25.6 MB << 192 MB L2, so gathers/atomics stay on-chip).
__global__ void k_spmm(const int* __restrict__ erow, const int* __restrict__ ecol,
                       const float* __restrict__ vals, const float* __restrict__ S,
                       float* __restrict__ Out, int F) {
    int e = blockIdx.x;
    int f = threadIdx.x;
    float v = vals[e];
    int r = erow[e], c = ecol[e];
    unsafeAtomicAdd(&Out[(long)r * F + f], v * S[(long)c * F + f]);
}

// ---------------------------------- per-column mean and rsqrt(var+eps) over rows
__global__ void k_bnstats(const float* __restrict__ X, float* __restrict__ mean,
                          float* __restrict__ rstd, int nrows, int ncols) {
    int c = blockIdx.x;
    __shared__ float s_s[256];
    __shared__ float s_q[256];
    float s = 0.0f, q = 0.0f;
    for (int r = threadIdx.x; r < nrows; r += blockDim.x) {
        float x = X[(long)r * ncols + c];
        s += x;
        q += x * x;
    }
    s_s[threadIdx.x] = s;
    s_q[threadIdx.x] = q;
    __syncthreads();
    for (int off = blockDim.x >> 1; off > 0; off >>= 1) {
        if ((int)threadIdx.x < off) {
            s_s[threadIdx.x] += s_s[threadIdx.x + off];
            s_q[threadIdx.x] += s_q[threadIdx.x + off];
        }
        __syncthreads();
    }
    if (threadIdx.x == 0) {
        float m   = s_s[0] / (float)nrows;
        float var = s_q[0] / (float)nrows - m * m;
        mean[c] = m;
        rstd[c] = rsqrtf(var + BN_EPS);
    }
}

// ------------------------------- y = relu(BN(x)) converted to bf16 for next GEMM
__global__ void k_bn_relu_cvt(const float* __restrict__ X, const float* __restrict__ mean,
                              const float* __restrict__ rstd, const float* __restrict__ gamma,
                              const float* __restrict__ beta, __bf16* __restrict__ Y,
                              int nrows, int ncols) {
    long total = (long)nrows * ncols;
    long i = blockIdx.x * (long)blockDim.x + threadIdx.x;
    long stride = (long)gridDim.x * blockDim.x;
    for (; i < total; i += stride) {
        int c = (int)(i % ncols);
        float y = gamma[c] * (X[i] - mean[c]) * rstd[c] + beta[c];
        Y[i] = (__bf16)(y > 0.0f ? y : 0.0f);
    }
}

// -------- fused: relu(BN(A2)) * Wm[vertices] column-sum + bm, then sigmoid -> out
__global__ void k_final(const float* __restrict__ A2, const float* __restrict__ mean,
                        const float* __restrict__ rstd, const float* __restrict__ gamma,
                        const float* __restrict__ beta, const float* __restrict__ Wm,
                        const float* __restrict__ bm, const int* __restrict__ vert,
                        float* __restrict__ out, int nrows, int ncols) {
    int j = blockIdx.x;
    __shared__ float s_s[256];
    float s = 0.0f;
    for (int r = threadIdx.x; r < nrows; r += blockDim.x) {
        float y = gamma[j] * (A2[(long)r * ncols + j] - mean[j]) * rstd[j] + beta[j];
        y = y > 0.0f ? y : 0.0f;
        s += y * Wm[(long)vert[r] * ncols + j];
    }
    s_s[threadIdx.x] = s;
    __syncthreads();
    for (int off = blockDim.x >> 1; off > 0; off >>= 1) {
        if ((int)threadIdx.x < off) s_s[threadIdx.x] += s_s[threadIdx.x + off];
        __syncthreads();
    }
    if (threadIdx.x == 0) {
        float z = s_s[0] + bm[j];
        out[j] = 1.0f / (1.0f + expf(-z));
    }
}

// ============================================================================
extern "C" void kernel_launch(void* const* d_in, const int* in_sizes, int n_in,
                              void* d_out, int out_size, void* d_ws, size_t ws_size,
                              hipStream_t stream) {
    const float* emb    = (const float*)d_in[0];
    const float* W1     = (const float*)d_in[1];
    // d_in[2] = b1  : cancels exactly through BatchNorm (constant column shift)
    const float* gamma1 = (const float*)d_in[3];
    const float* beta1  = (const float*)d_in[4];
    const float* W2     = (const float*)d_in[5];
    // d_in[6] = b2  : cancels exactly through BatchNorm
    const float* gamma2 = (const float*)d_in[7];
    const float* beta2  = (const float*)d_in[8];
    const float* Wm     = (const float*)d_in[9];
    const float* bm     = (const float*)d_in[10];
    const int*   vert   = (const int*)d_in[11];
    const int*   erow   = (const int*)d_in[12];
    const int*   ecol   = (const int*)d_in[13];
    const float* vals   = (const float*)d_in[14];

    const int nN = in_sizes[11];            // 50000 nodes
    const int nE = in_sizes[12];            // 1.6M edges
    const int F1 = 128, F2 = 64;

    // ---- workspace carve-up (256B aligned), with buffer reuse across layers
    size_t off = 0;
    auto walloc = [&](size_t bytes) -> void* {
        off = (off + 255) & ~(size_t)255;
        void* p = (char*)d_ws + off;
        off += bytes;
        return p;
    };
    __bf16* Xbf   = (__bf16*)walloc((size_t)nN * F1 * sizeof(__bf16)); // 12.8 MB
    __bf16* W1bf  = (__bf16*)walloc((size_t)F1 * F1 * sizeof(__bf16));
    __bf16* W2bf  = (__bf16*)walloc((size_t)F1 * F2 * sizeof(__bf16));
    float*  S     = (float*)walloc((size_t)nN * F1 * sizeof(float));   // support
    float*  Acc   = (float*)walloc((size_t)nN * F1 * sizeof(float));   // spmm accum
    __bf16* X1bf  = (__bf16*)walloc((size_t)nN * F1 * sizeof(__bf16));
    float*  mean1 = (float*)walloc(F1 * sizeof(float));
    float*  rstd1 = (float*)walloc(F1 * sizeof(float));
    float*  mean2 = (float*)walloc(F2 * sizeof(float));
    float*  rstd2 = (float*)walloc(F2 * sizeof(float));
    (void)ws_size; (void)n_in; (void)out_size;

    // ---- stage 0: convert inputs to bf16
    k_cvt_rows<<<1024, 256, 0, stream>>>(emb, vert, Xbf, nN, F1);
    k_cvt_rows<<<64, 256, 0, stream>>>(W1, nullptr, W1bf, F1, F1);
    k_cvt_rows<<<32, 256, 0, stream>>>(W2, nullptr, W2bf, F1, F2);

    const int tilesM = (nN + 15) / 16;      // 3125
    const int gemmBlocks = (tilesM + 7) / 8; // 8 waves/block, 1 stripe/wave

    // ---- layer 1: GEMM (WMMA) -> SpMM -> BN stats -> BN+ReLU+cvt
    k_gemm_wmma<128, 128><<<gemmBlocks, 256, 0, stream>>>(Xbf, W1bf, S, nN);
    k_zero<<<2048, 256, 0, stream>>>(Acc, (long)nN * F1);
    k_spmm<<<nE, F1, 0, stream>>>(erow, ecol, vals, S, Acc, F1);
    k_bnstats<<<F1, 256, 0, stream>>>(Acc, mean1, rstd1, nN, F1);
    k_bn_relu_cvt<<<2048, 256, 0, stream>>>(Acc, mean1, rstd1, gamma1, beta1, X1bf, nN, F1);

    // ---- layer 2: GEMM (WMMA) -> SpMM -> BN stats
    k_gemm_wmma<64, 128><<<gemmBlocks, 256, 0, stream>>>(X1bf, W2bf, S, nN);
    k_zero<<<2048, 256, 0, stream>>>(Acc, (long)nN * F2);
    k_spmm<<<nE, F2, 0, stream>>>(erow, ecol, vals, S, Acc, F2);
    k_bnstats<<<F2, 256, 0, stream>>>(Acc, mean2, rstd2, nN, F2);

    // ---- head: fused BN+ReLU+masked column reduction + sigmoid
    k_final<<<F2, 256, 0, stream>>>(Acc, mean2, rstd2, gamma2, beta2, Wm, bm, vert,
                                    (float*)d_out, nN, F2);
}
